// DisGeneFormer_17162689315199
// MI455X (gfx1250) — matrix operand order repaired and checked by reading, not compiled, not run on hardware
//
#include <hip/hip_runtime.h>
#include <math.h>

// ---------------------------------------------------------------------------
// Types for CDNA5 WMMA (wave32): v_wmma_f32_16x16x32_bf16
// ---------------------------------------------------------------------------
typedef __attribute__((ext_vector_type(16))) __bf16 v16bf;
typedef __attribute__((ext_vector_type(8)))  __bf16 v8bf;
typedef __attribute__((ext_vector_type(8)))  float  v8f;

static inline unsigned cdiv(unsigned a, unsigned b) { return (a + b - 1) / b; }

__device__ __forceinline__ float apply_act(float v, int act) {
  if (act == 1) return v > 0.f ? v : 0.01f * v;   // leaky_relu(0.01)
  if (act == 2) return fmaxf(v, 0.f);             // relu
  return v;
}

__device__ __forceinline__ v8bf pack8(float4 f0, float4 f1) {
  v8bf pk;
  pk[0] = (__bf16)f0.x; pk[1] = (__bf16)f0.y;
  pk[2] = (__bf16)f0.z; pk[3] = (__bf16)f0.w;
  pk[4] = (__bf16)f1.x; pk[5] = (__bf16)f1.y;
  pk[6] = (__bf16)f1.z; pk[7] = (__bf16)f1.w;
  return pk;
}

// ---------------------------------------------------------------------------
// Generic GEMM: C[M,N] = act(A[M,K] @ B[N,K]^T + bias[N])
// Block: 256 threads (8 waves). Block tile 128x64; each wave owns a 32x32
// output patch -> 4 x v_wmma_f32_16x16x32_bf16 per K-step.
// Fast variant (whole tile in bounds, K%32==0): branch-free K loop with
// register-prefetch double buffering; float4 loads + 16B packed-bf16 LDS
// stores. Generic guarded variant only for edge tiles.
// ---------------------------------------------------------------------------
#define GT_M 128
#define GT_N 64
#define GT_K 32
#define LDP  40   // padded LDS row stride (bf16 elems): 80B rows keep 16B align

__global__ __launch_bounds__(256) void gemm_bf16_wmma(
    const float* __restrict__ A, const float* __restrict__ B,
    const float* __restrict__ bias, float* __restrict__ C,
    int M, int N, int K, int act)
{
  __shared__ __bf16 sA[GT_M][LDP];
  __shared__ __bf16 sB[GT_N][LDP];

  const int tid  = threadIdx.x;
  const int wave = tid >> 5;
  const int lane = tid & 31;

  const int bm = blockIdx.y * GT_M;
  const int bn = blockIdx.x * GT_N;

  const int wm = (wave & 3) * 32;   // wave M offset (two 16-row tiles)
  const int wn = (wave >> 2) * 32;  // wave N offset (two 16-col tiles)

  v8f acc00 = {}, acc01 = {}, acc10 = {}, acc11 = {};

  const int row = lane & 15;
  const int kA  = (lane < 16) ? 0 : 8;   // A frag: lanes0-15 K{0-7,16-23}, 16-31 {8-15,24-31}
  const int kB  = (lane < 16) ? 0 : 8;   // B frag: lanes0-15 K{0-7,16-23}, 16-31 {8-15,24-31}

  // staging coords: thread -> (r, c) with r = tid/4, c in {0,8,16,24}
  const int sr = tid >> 2;
  const int sc = (tid & 3) * 8;

  const bool fast = (bm + GT_M <= M) && (bn + GT_N <= N) && ((K & (GT_K - 1)) == 0);

  if (fast) {
    const float* ap0 = A + (size_t)(bm + sr) * K + sc;        // rows sr, sr+64
    const float* ap1 = A + (size_t)(bm + sr + 64) * K + sc;
    const float* bp  = B + (size_t)(bn + sr) * K + sc;

    // preload K-step 0 into registers
    float4 pa0 = *(const float4*)(ap0);
    float4 pa1 = *(const float4*)(ap0 + 4);
    float4 pa2 = *(const float4*)(ap1);
    float4 pa3 = *(const float4*)(ap1 + 4);
    float4 pb0 = *(const float4*)(bp);
    float4 pb1 = *(const float4*)(bp + 4);

    for (int k0 = 0; k0 < K; k0 += GT_K) {
      // commit prefetched tile to LDS
      *(v8bf*)&sA[sr][sc]      = pack8(pa0, pa1);
      *(v8bf*)&sA[sr + 64][sc] = pack8(pa2, pa3);
      *(v8bf*)&sB[sr][sc]      = pack8(pb0, pb1);
      __syncthreads();

      // prefetch next K-step while WMMAs run off LDS
      if (k0 + GT_K < K) {
        ap0 += GT_K; ap1 += GT_K; bp += GT_K;
        pa0 = *(const float4*)(ap0);
        pa1 = *(const float4*)(ap0 + 4);
        pa2 = *(const float4*)(ap1);
        pa3 = *(const float4*)(ap1 + 4);
        pb0 = *(const float4*)(bp);
        pb1 = *(const float4*)(bp + 4);
      }

      v16bf af0, af1, b0, b1;
#pragma unroll
      for (int i = 0; i < 8; ++i) {
        af0[i]     = sA[wm + row][kA + i];
        af0[8 + i] = sA[wm + row][16 + kA + i];
        af1[i]     = sA[wm + 16 + row][kA + i];
        af1[8 + i] = sA[wm + 16 + row][16 + kA + i];
      }
#pragma unroll
      for (int i = 0; i < 16; ++i) {
        b0[i] = sB[wn + row][kB + (i & 7) + ((i >> 3) << 4)];
        b1[i] = sB[wn + 16 + row][kB + (i & 7) + ((i >> 3) << 4)];
      }

      acc00 = __builtin_amdgcn_wmma_f32_16x16x32_bf16(false, af0, false, b0,
                                                      (short)0, acc00, false, false);
      acc01 = __builtin_amdgcn_wmma_f32_16x16x32_bf16(false, af0, false, b1,
                                                      (short)0, acc01, false, false);
      acc10 = __builtin_amdgcn_wmma_f32_16x16x32_bf16(false, af1, false, b0,
                                                      (short)0, acc10, false, false);
      acc11 = __builtin_amdgcn_wmma_f32_16x16x32_bf16(false, af1, false, b1,
                                                      (short)0, acc11, false, false);
      __syncthreads();
    }
  } else {
    for (int k0 = 0; k0 < K; k0 += GT_K) {
      for (int i = tid; i < GT_M * GT_K; i += 256) {
        int r = i >> 5, c = i & 31;
        int gr = bm + r, gc = k0 + c;
        float v = (gr < M && gc < K) ? A[(size_t)gr * K + gc] : 0.f;
        sA[r][c] = (__bf16)v;
      }
      for (int i = tid; i < GT_N * GT_K; i += 256) {
        int r = i >> 5, c = i & 31;
        int gr = bn + r, gc = k0 + c;
        float v = (gr < N && gc < K) ? B[(size_t)gr * K + gc] : 0.f;
        sB[r][c] = (__bf16)v;
      }
      __syncthreads();

      v16bf af0, af1, b0, b1;
#pragma unroll
      for (int i = 0; i < 8; ++i) {
        af0[i]     = sA[wm + row][kA + i];
        af0[8 + i] = sA[wm + row][16 + kA + i];
        af1[i]     = sA[wm + 16 + row][kA + i];
        af1[8 + i] = sA[wm + 16 + row][16 + kA + i];
      }
#pragma unroll
      for (int i = 0; i < 16; ++i) {
        b0[i] = sB[wn + row][kB + (i & 7) + ((i >> 3) << 4)];
        b1[i] = sB[wn + 16 + row][kB + (i & 7) + ((i >> 3) << 4)];
      }

      acc00 = __builtin_amdgcn_wmma_f32_16x16x32_bf16(false, af0, false, b0,
                                                      (short)0, acc00, false, false);
      acc01 = __builtin_amdgcn_wmma_f32_16x16x32_bf16(false, af0, false, b1,
                                                      (short)0, acc01, false, false);
      acc10 = __builtin_amdgcn_wmma_f32_16x16x32_bf16(false, af1, false, b0,
                                                      (short)0, acc10, false, false);
      acc11 = __builtin_amdgcn_wmma_f32_16x16x32_bf16(false, af1, false, b1,
                                                      (short)0, acc11, false, false);
      __syncthreads();
    }
  }

  // store: VGPR j -> row (j or 8+j), col = lane%16
  const int col   = lane & 15;
  const int rbase = (lane < 16) ? 0 : 8;
#pragma unroll
  for (int j = 0; j < 8; ++j) {
    int gr0 = bm + wm + rbase + j;
    int gr1 = gr0 + 16;
    int gc0 = bn + wn + col;
    int gc1 = gc0 + 16;
    float bv0 = (bias && gc0 < N) ? bias[gc0] : 0.f;
    float bv1 = (bias && gc1 < N) ? bias[gc1] : 0.f;
    if (gr0 < M) {
      if (gc0 < N) C[(size_t)gr0 * N + gc0] = apply_act(acc00[j] + bv0, act);
      if (gc1 < N) C[(size_t)gr0 * N + gc1] = apply_act(acc01[j] + bv1, act);
    }
    if (gr1 < M) {
      if (gc0 < N) C[(size_t)gr1 * N + gc0] = apply_act(acc10[j] + bv0, act);
      if (gc1 < N) C[(size_t)gr1 * N + gc1] = apply_act(acc11[j] + bv1, act);
    }
  }
}

// ---------------------------------------------------------------------------
// Utility kernels
// ---------------------------------------------------------------------------
__global__ void fill_f32(float* p, float v, long n) {
  long i = (long)blockIdx.x * blockDim.x + threadIdx.x;
  if (i < n) p[i] = v;
}

__device__ __forceinline__ void atomic_max_f32(float* addr, float value) {
  if (value >= 0.f) atomicMax((int*)addr, __float_as_int(value));
  else              atomicMin((unsigned int*)addr, __float_as_uint(value));
}

__device__ __forceinline__ void edge_nodes(const int* __restrict__ ei, int E,
                                           int e, int& s, int& d) {
  if (e < E) { s = ei[e]; d = ei[E + e]; }
  else       { s = e - E; d = e - E; }     // self loops appended
}

// a_src[n,h] = sum_c xl[n,h,c]*att_src[h,c]; same for dst
__global__ void gat_logits(const float* __restrict__ xl,
                           const float* __restrict__ att_src,
                           const float* __restrict__ att_dst,
                           float* __restrict__ asrc, float* __restrict__ adst,
                           int N, int H, int C) {
  int t = blockIdx.x * blockDim.x + threadIdx.x;
  if (t >= N * H) return;
  int n = t / H, h = t % H;
  const float* x = xl + (size_t)n * H * C + (size_t)h * C;
  float ss = 0.f, sd = 0.f;
  for (int c = 0; c < C; ++c) { ss += x[c] * att_src[h * C + c]; sd += x[c] * att_dst[h * C + c]; }
  asrc[t] = ss; adst[t] = sd;
}

__global__ void edge_max(const int* __restrict__ ei, int E, int N,
                         const float* __restrict__ asrc, const float* __restrict__ adst,
                         float* __restrict__ m, int H) {
  int t = blockIdx.x * blockDim.x + threadIdx.x;
  int ET = E + N;
  if (t >= ET * H) return;
  int e = t / H, h = t % H, s, d;
  edge_nodes(ei, E, e, s, d);
  float a = asrc[s * H + h] + adst[d * H + h];
  a = a > 0.f ? a : 0.2f * a;
  atomic_max_f32(&m[d * H + h], a);
}

__global__ void edge_expsum(const int* __restrict__ ei, int E, int N,
                            const float* __restrict__ asrc, const float* __restrict__ adst,
                            const float* __restrict__ m, float* __restrict__ ssum, int H) {
  int t = blockIdx.x * blockDim.x + threadIdx.x;
  int ET = E + N;
  if (t >= ET * H) return;
  int e = t / H, h = t % H, s, d;
  edge_nodes(ei, E, e, s, d);
  float a = asrc[s * H + h] + adst[d * H + h];
  a = a > 0.f ? a : 0.2f * a;
  atomicAdd(&ssum[d * H + h], __expf(a - m[d * H + h]));
}

// out[dst,h,c] += xl[src,h,c] * alpha(e,h)   -- one block per edge
__global__ void edge_agg(const int* __restrict__ ei, int E, int N,
                         const float* __restrict__ xl,
                         const float* __restrict__ asrc, const float* __restrict__ adst,
                         const float* __restrict__ m, const float* __restrict__ ssum,
                         float* __restrict__ out, int H, int C) {
  int e = blockIdx.x, s, d;
  edge_nodes(ei, E, e, s, d);
  int HC = H * C;
  for (int i = threadIdx.x; i < HC; i += blockDim.x) {
    int h = i / C;
    float a = asrc[s * H + h] + adst[d * H + h];
    a = a > 0.f ? a : 0.2f * a;
    float alpha = __expf(a - m[d * H + h]) / ssum[d * H + h];
    atomicAdd(&out[(size_t)d * HC + i], xl[(size_t)s * HC + i] * alpha);
  }
}

// BatchNorm stats over N rows of lrelu(x + bias) : one block per channel
__global__ __launch_bounds__(256) void bn_stats(const float* __restrict__ x,
                                                const float* __restrict__ bias,
                                                float* __restrict__ mu, float* __restrict__ rstd,
                                                int N, int Cn) {
  int c = blockIdx.x, tid = threadIdx.x;
  float s = 0.f, s2 = 0.f;
  for (int n = tid; n < N; n += 256) {
    float v = x[(size_t)n * Cn + c] + bias[c];
    v = v > 0.f ? v : 0.01f * v;
    s += v; s2 += v * v;
  }
  __shared__ float rs[256], rq[256];
  rs[tid] = s; rq[tid] = s2; __syncthreads();
  for (int o = 128; o > 0; o >>= 1) {
    if (tid < o) { rs[tid] += rs[tid + o]; rq[tid] += rq[tid + o]; }
    __syncthreads();
  }
  if (tid == 0) {
    float mean = rs[0] / (float)N;
    float var  = rq[0] / (float)N - mean * mean;
    mu[c] = mean;
    rstd[c] = rsqrtf(var + 1e-5f);
  }
}

__global__ void bn_apply(const float* __restrict__ x, const float* __restrict__ bias,
                         const float* __restrict__ mu, const float* __restrict__ rstd,
                         const float* __restrict__ g, const float* __restrict__ b,
                         float* __restrict__ out, long total, int Cn) {
  long t = (long)blockIdx.x * blockDim.x + threadIdx.x;
  if (t >= total) return;
  int c = (int)(t % Cn);
  float v = x[t] + bias[c];
  v = v > 0.f ? v : 0.01f * v;
  out[t] = (v - mu[c]) * rstd[c] * g[c] + b[c];
}

// 3-way attention aggregation across layer axis: one thread per (node, head)
__global__ void agg_attn(const float* __restrict__ qkv, float* __restrict__ o,
                         int N, int D, int H) {
  int t = blockIdx.x * blockDim.x + threadIdx.x;
  if (t >= N * H) return;
  int n = t / H, h = t % H;
  int dh = D / H;
  const float scale = rsqrtf((float)dh);
  float s[3][3];
  for (int l = 0; l < 3; ++l)
    for (int mm = 0; mm < 3; ++mm) {
      const float* qr = qkv + (size_t)(l * N + n) * 3 * D + h * dh;
      const float* kr = qkv + (size_t)(mm * N + n) * 3 * D + D + h * dh;
      float acc = 0.f;
      for (int dd = 0; dd < dh; ++dd) acc += qr[dd] * kr[dd];
      s[l][mm] = acc * scale;
    }
  for (int l = 0; l < 3; ++l) {
    float mx = fmaxf(s[l][0], fmaxf(s[l][1], s[l][2]));
    float e0 = __expf(s[l][0] - mx), e1 = __expf(s[l][1] - mx), e2 = __expf(s[l][2] - mx);
    float inv = 1.f / (e0 + e1 + e2);
    s[l][0] = e0 * inv; s[l][1] = e1 * inv; s[l][2] = e2 * inv;
  }
  for (int dd = 0; dd < dh; ++dd) {
    float v0 = qkv[(size_t)(0 * N + n) * 3 * D + 2 * D + h * dh + dd];
    float v1 = qkv[(size_t)(1 * N + n) * 3 * D + 2 * D + h * dh + dd];
    float v2 = qkv[(size_t)(2 * N + n) * 3 * D + 2 * D + h * dh + dd];
    for (int l = 0; l < 3; ++l)
      o[(size_t)(l * N + n) * D + h * dh + dd] = s[l][0] * v0 + s[l][1] * v1 + s[l][2] * v2;
  }
}

__global__ void agg_combine(const float* __restrict__ oproj, const float* __restrict__ alpha,
                            float* __restrict__ zout, int N, int D) {
  long t = (long)blockIdx.x * blockDim.x + threadIdx.x;
  if (t >= (long)N * D) return;
  float a0 = alpha[0], a1 = alpha[1], a2 = alpha[2];
  float mx = fmaxf(a0, fmaxf(a1, a2));
  float e0 = __expf(a0 - mx), e1 = __expf(a1 - mx), e2 = __expf(a2 - mx);
  float inv = 1.f / (e0 + e1 + e2);
  int n = (int)(t / D), dd = (int)(t % D);
  zout[t] = (e0 * oproj[(size_t)(0 * N + n) * D + dd] +
             e1 * oproj[(size_t)(1 * N + n) * D + dd] +
             e2 * oproj[(size_t)(2 * N + n) * D + dd]) * inv;
}

// Flash-style self attention: one wave per (query, head); dh = 128 -> 4 dims/lane
__global__ __launch_bounds__(256) void mha_flash(const float* __restrict__ qkv,
                                                 float* __restrict__ out,
                                                 int S, int H, int dh) {
  int wg   = blockIdx.x * 8 + (threadIdx.x >> 5);
  int lane = threadIdx.x & 31;
  int q = wg / H, h = wg % H;
  if (q >= S) return;
  int D = H * dh;
  const float scale = rsqrtf((float)dh);
  const float* qrow = qkv + (size_t)q * 3 * D + h * dh + lane * 4;
  float q0 = qrow[0] * scale, q1 = qrow[1] * scale, q2 = qrow[2] * scale, q3 = qrow[3] * scale;
  float m = -INFINITY, den = 0.f;
  float a0 = 0.f, a1 = 0.f, a2 = 0.f, a3 = 0.f;
  for (int k = 0; k < S; ++k) {
    const float4 kv = *(const float4*)(qkv + (size_t)k * 3 * D + D + h * dh + lane * 4);
    float p = q0 * kv.x + q1 * kv.y + q2 * kv.z + q3 * kv.w;
#pragma unroll
    for (int off = 16; off > 0; off >>= 1) p += __shfl_xor(p, off, 32);
    float nm = fmaxf(m, p);
    float ex = __expf(p - nm);
    float sc = __expf(m - nm);
    const float4 vv = *(const float4*)(qkv + (size_t)k * 3 * D + 2 * D + h * dh + lane * 4);
    a0 = a0 * sc + ex * vv.x;
    a1 = a1 * sc + ex * vv.y;
    a2 = a2 * sc + ex * vv.z;
    a3 = a3 * sc + ex * vv.w;
    den = den * sc + ex;
    m = nm;
  }
  float inv = 1.f / den;
  float* orow = out + (size_t)q * D + h * dh + lane * 4;
  orow[0] = a0 * inv; orow[1] = a1 * inv; orow[2] = a2 * inv; orow[3] = a3 * inv;
}

// out = LayerNorm(x + y) with D = 512 (one block of 256 per row)
__global__ __launch_bounds__(256) void add_ln_512(const float* __restrict__ x,
                                                  const float* __restrict__ y,
                                                  const float* __restrict__ g,
                                                  const float* __restrict__ b,
                                                  float* __restrict__ out) {
  int row = blockIdx.x, tid = threadIdx.x;
  size_t base = (size_t)row * 512;
  float v0 = x[base + tid] + y[base + tid];
  float v1 = x[base + tid + 256] + y[base + tid + 256];
  __shared__ float red[256];
  __shared__ float mu_s, rs_s;
  red[tid] = v0 + v1; __syncthreads();
  for (int o = 128; o > 0; o >>= 1) { if (tid < o) red[tid] += red[tid + o]; __syncthreads(); }
  if (tid == 0) mu_s = red[0] * (1.f / 512.f);
  __syncthreads();
  float mu = mu_s;
  float d0 = v0 - mu, d1 = v1 - mu;
  red[tid] = d0 * d0 + d1 * d1; __syncthreads();
  for (int o = 128; o > 0; o >>= 1) { if (tid < o) red[tid] += red[tid + o]; __syncthreads(); }
  if (tid == 0) rs_s = rsqrtf(red[0] * (1.f / 512.f) + 1e-5f);
  __syncthreads();
  float rs = rs_s;
  out[base + tid]       = d0 * rs * g[tid] + b[tid];
  out[base + tid + 256] = d1 * rs * g[tid + 256] + b[tid + 256];
}

__global__ void gather_concat(const float* __restrict__ z, const int* __restrict__ bidx,
                              float* __restrict__ out, int B, int NGr, int D) {
  long t = (long)blockIdx.x * blockDim.x + threadIdx.x;
  if (t >= (long)B * 2 * D) return;
  int i = (int)(t / (2 * D));
  int j = (int)(t % (2 * D));
  int srow = (j < D) ? bidx[i * 2 + 0] : (NGr + bidx[i * 2 + 1]);
  int col  = (j < D) ? j : j - D;
  out[t] = z[(size_t)srow * D + col];
}

// ---------------------------------------------------------------------------
// Host orchestration
// ---------------------------------------------------------------------------
extern "C" void kernel_launch(void* const* d_in, const int* in_sizes, int n_in,
                              void* d_out, int out_size, void* d_ws, size_t ws_size,
                              hipStream_t stream) {
  constexpr int NG = 2400, ND = 1200, EG = 24000, ED = 12000;
  constexpr int FDG = 1024, FDD = 768, HID = 512, B = 1024, S = NG + ND;

  auto F = [&](int i) { return (const float*)d_in[i]; };
  auto Ii = [&](int i) { return (const int*)d_in[i]; };

  // --- workspace carve (floats) ---
  float* w = (float*)d_ws;
  size_t off = 0;
  auto alloc = [&](size_t n) { float* p = w + off; off += n; return p; };
  float* XL   = alloc((size_t)NG * 2048);        // GAT linear outputs / FC scratch
  float* GOUT = alloc((size_t)NG * 2048);        // edge-aggregate accumulator
  float* HA   = alloc((size_t)NG * 2048);        // h0 / FC scratch
  float* HB   = alloc((size_t)NG * 2048);        // h1
  float* STK  = alloc((size_t)3 * NG * HID);     // stacked [h0p,h1p,h2]
  float* QKV  = alloc((size_t)3 * NG * 3 * HID); // qkv / FFN mid (11M floats)
  float* OAGG = alloc((size_t)3 * NG * HID);
  float* OPRJ = alloc((size_t)3 * NG * HID);
  float* Z    = alloc((size_t)S * HID);
  float* ZT   = alloc((size_t)S * HID);
  float* ASRC = alloc((size_t)NG * 4);
  float* ADST = alloc((size_t)NG * 4);
  float* MBUF = alloc((size_t)NG * 4);
  float* SBUF = alloc((size_t)NG * 4);
  float* MU   = alloc(2048);
  float* RSTD = alloc(2048);

  auto gemm = [&](const float* A, const float* Bp, const float* bias, float* Cp,
                  int M, int Nc, int K, int act) {
    dim3 g(cdiv(Nc, GT_N), cdiv(M, GT_M));
    gemm_bf16_wmma<<<g, 256, 0, stream>>>(A, Bp, bias, Cp, M, Nc, K, act);
  };
  auto fill = [&](float* p, float v, long n) {
    fill_f32<<<cdiv((unsigned)n, 256), 256, 0, stream>>>(p, v, n);
  };

  // One GATConv stage: xl -> (attention scatter) -> lrelu(+bias) -> BN -> hout
  auto gat_stage = [&](const float* xl, const float* att_src, const float* att_dst,
                       const float* convb, const float* bng, const float* bnb,
                       int Nn, int Ne, const int* ei, int H, int C, float* hout) {
    int NH = Nn * H, ET = Ne + Nn;
    long tot = (long)Nn * H * C;
    gat_logits<<<cdiv(NH, 256), 256, 0, stream>>>(xl, att_src, att_dst, ASRC, ADST, Nn, H, C);
    fill(MBUF, -INFINITY, NH);
    fill(SBUF, 0.f, NH);
    fill(GOUT, 0.f, tot);
    edge_max<<<cdiv(ET * H, 256), 256, 0, stream>>>(ei, Ne, Nn, ASRC, ADST, MBUF, H);
    edge_expsum<<<cdiv(ET * H, 256), 256, 0, stream>>>(ei, Ne, Nn, ASRC, ADST, MBUF, SBUF, H);
    edge_agg<<<ET, 256, 0, stream>>>(ei, Ne, Nn, xl, ASRC, ADST, MBUF, SBUF, GOUT, H, C);
    bn_stats<<<H * C, 256, 0, stream>>>(GOUT, convb, MU, RSTD, Nn, H * C);
    bn_apply<<<cdiv((unsigned)tot, 256), 256, 0, stream>>>(GOUT, convb, MU, RSTD, bng, bnb,
                                                           hout, tot, H * C);
  };

  // pb = base index of graphp params (alphabetical pytree flatten):
  // +0 agg.Wo +1 agg.Wqkv +2 agg.alpha +3 agg.bo +4 agg.bqkv
  // +5 bn0.b +6 bn0.g +7 bn1.b +8 bn1.g +9 bn2.b +10 bn2.g
  // +11..14 conv0{W,att_dst,att_src,b} +15..18 conv1 +19..22 conv2
  // +23 proj0.W +24 proj0.b +25 proj1.W +26 proj1.b
  auto run_branch = [&](const float* x, int din, int Nn, int Ne, const int* ei,
                        int pb, float* zdst) {
    const float *aggWo = F(pb + 0), *aggWqkv = F(pb + 1), *aggAl = F(pb + 2);
    const float *aggBo = F(pb + 3), *aggBqkv = F(pb + 4);
    const float *bnb0 = F(pb + 5), *bng0 = F(pb + 6);
    const float *bnb1 = F(pb + 7), *bng1 = F(pb + 8);
    const float *bnb2 = F(pb + 9), *bng2 = F(pb + 10);
    const float *W0 = F(pb + 11), *ad0 = F(pb + 12), *as0 = F(pb + 13), *cb0 = F(pb + 14);
    const float *W1 = F(pb + 15), *ad1 = F(pb + 16), *as1 = F(pb + 17), *cb1 = F(pb + 18);
    const float *W2 = F(pb + 19), *ad2 = F(pb + 20), *as2 = F(pb + 21), *cb2 = F(pb + 22);
    const float *p0W = F(pb + 23), *p0b = F(pb + 24);
    const float *p1W = F(pb + 25), *p1b = F(pb + 26);

    // conv0 (4 heads x 512)
    gemm(x, W0, nullptr, XL, Nn, 2048, din, 0);
    gat_stage(XL, as0, ad0, cb0, bng0, bnb0, Nn, Ne, ei, 4, HID, HA);
    gemm(HA, p0W, p0b, STK, Nn, HID, 2048, 0);                       // h0p
    // conv1
    gemm(HA, W1, nullptr, XL, Nn, 2048, 2048, 0);
    gat_stage(XL, as1, ad1, cb1, bng1, bnb1, Nn, Ne, ei, 4, HID, HB);
    gemm(HB, p1W, p1b, STK + (size_t)Nn * HID, Nn, HID, 2048, 0);    // h1p
    // conv2 (1 head x 512)
    gemm(HB, W2, nullptr, XL, Nn, HID, 2048, 0);
    gat_stage(XL, as2, ad2, cb2, bng2, bnb2, Nn, Ne, ei, 1, HID,
              STK + (size_t)2 * Nn * HID);                            // h2
    // attention aggregation over the 3 stacked layers
    gemm(STK, aggWqkv, aggBqkv, QKV, 3 * Nn, 3 * HID, HID, 0);
    agg_attn<<<cdiv(Nn * 4, 256), 256, 0, stream>>>(QKV, OAGG, Nn, HID, 4);
    gemm(OAGG, aggWo, aggBo, OPRJ, 3 * Nn, HID, HID, 0);
    agg_combine<<<cdiv(Nn * HID, 256), 256, 0, stream>>>(OPRJ, aggAl, zdst, Nn, HID);
  };

  // --- input index map (alphabetical pytree flatten) ---
  const int* batch_idx = Ii(0);
  const int* d_ei      = Ii(1);
  const float* dis_x   = F(29);
  const int* g_ei      = Ii(36);
  const float* gen_x   = F(64);

  // --- GAT branches -> Z ---
  run_branch(gen_x, FDG, NG, EG, g_ei, 37, Z);
  run_branch(dis_x, FDD, ND, ED, d_ei, 2, Z + (size_t)NG * HID);

  // --- transformer encoder (2 layers, post-norm) ---
  for (int l = 0; l < 2; ++l) {
    int tb = 65 + 12 * l;   // W1 W2 Wo Wqkv b1 b2 bo bqkv ln1_b ln1_g ln2_b ln2_g
    const float *tW1 = F(tb + 0), *tW2 = F(tb + 1), *tWo = F(tb + 2), *tWqkv = F(tb + 3);
    const float *tb1 = F(tb + 4), *tb2 = F(tb + 5), *tbo = F(tb + 6), *tbqkv = F(tb + 7);
    const float *ln1b = F(tb + 8), *ln1g = F(tb + 9), *ln2b = F(tb + 10), *ln2g = F(tb + 11);

    gemm(Z, tWqkv, tbqkv, QKV, S, 3 * HID, HID, 0);
    mha_flash<<<cdiv(S * 4, 8), 256, 0, stream>>>(QKV, ZT, S, 4, 128);
    gemm(ZT, tWo, tbo, OPRJ, S, HID, HID, 0);
    add_ln_512<<<S, 256, 0, stream>>>(Z, OPRJ, ln1g, ln1b, Z);
    gemm(Z, tW1, tb1, QKV, S, 2048, HID, 2);          // relu FFN
    gemm(QKV, tW2, tb2, ZT, S, HID, 2048, 0);
    add_ln_512<<<S, 256, 0, stream>>>(Z, ZT, ln2g, ln2b, Z);
  }

  // --- FC head ---
  const float *fW1 = F(30), *fW2 = F(31), *fW3 = F(32);
  const float *fb1 = F(33), *fb2 = F(34), *fb3 = F(35);
  gather_concat<<<cdiv(B * 1024, 256), 256, 0, stream>>>(Z, batch_idx, XL, B, NG, HID);
  gemm(XL, fW1, fb1, GOUT, B, 2048, 1024, 1);         // leaky_relu 0.01
  gemm(GOUT, fW2, fb2, HA, B, 1024, 2048, 1);
  gemm(HA, fW3, fb3, (float*)d_out, B, 2, 1024, 0);
}
